// TokenToPatchMaxSimAligner_31628139168208
// MI455X (gfx1250) — matrix operand order repaired and checked by reading, not compiled, not run on hardware
//
#include <hip/hip_runtime.h>
#include <math.h>

typedef __attribute__((ext_vector_type(16))) _Float16 v16h;
typedef __attribute__((ext_vector_type(8)))  _Float16 v8h;
typedef __attribute__((ext_vector_type(4)))  _Float16 v4h;
typedef __attribute__((ext_vector_type(8)))  float    v8f;

#define B 128
#define M 256
#define N 1024
#define D 256

__device__ __forceinline__ v8f wmma16(v16h a, v16h b, v8f c) {
  // D(16x16 f32) = A(16x32 f16) * B(32x16 f16) + C
  return __builtin_amdgcn_wmma_f32_16x16x32_f16(false, a, false, b, (short)0, c,
                                                false, false);
}

// Build A fragment from an f16 row (K-contiguous). half = lane>>4.
// v16h element j: K = 16*half + j (contiguous).
__device__ __forceinline__ v16h load_afrag(const _Float16* __restrict__ row, int half) {
  v8h lo = *(const v8h*)(row + 16 * half);
  v8h hi = *(const v8h*)(row + 16 * half + 8);
  return __builtin_shufflevector(lo, hi, 0, 1, 2, 3, 4, 5, 6, 7,
                                 8, 9, 10, 11, 12, 13, 14, 15);
}
// B fragment: lane = col n, halves K = [16*half, 16*half+16) contiguous.
__device__ __forceinline__ v16h load_bfrag(const _Float16* __restrict__ row, int half) {
  return *(const v16h*)(row + 16 * half);
}

__device__ __forceinline__ float gelu_exact(float x) {
  return 0.5f * x * (1.0f + erff(x * 0.70710678118654752f));
}

// ---------------------------------------------------------------------------
// Kernel 1: L2-normalize rows of length D=256, emit f16.
// ---------------------------------------------------------------------------
__global__ __launch_bounds__(256) void norm_rows_f16(const float* __restrict__ in,
                                                     _Float16* __restrict__ out) {
  const int row = blockIdx.x;
  const int t = threadIdx.x;
  float v = in[(size_t)row * D + t];
  __shared__ float red[256];
  red[t] = v * v;
  __syncthreads();
  for (int s = 128; s > 0; s >>= 1) {
    if (t < s) red[t] += red[t + s];
    __syncthreads();
  }
  float inv = 1.0f / fmaxf(sqrtf(red[0]), 1e-8f);
  out[(size_t)row * D + t] = (_Float16)(v * inv);
}

// ---------------------------------------------------------------------------
// Kernel 2: convert+transpose weights: out[n*K+k] = (f16) in[k*Nc+n].
// ---------------------------------------------------------------------------
__global__ __launch_bounds__(256) void conv_transpose_w(const float* __restrict__ in,
                                                        _Float16* __restrict__ out,
                                                        int K, int Nc) {
  int idx = blockIdx.x * 256 + threadIdx.x;
  int n = idx / K, k = idx % K;
  out[idx] = (_Float16)in[(size_t)k * Nc + n];
}

// ---------------------------------------------------------------------------
// Kernel 3: txtnT[b][d][m] = txtn_h[b][m][d], 32x32 LDS tiles.
// grid ((D/32)*(M/32), B), block 256.
// ---------------------------------------------------------------------------
__global__ __launch_bounds__(256) void transpose_f16(const _Float16* __restrict__ in,
                                                     _Float16* __restrict__ out) {
  const int b = blockIdx.y;
  const int td = (blockIdx.x % (D / 32)) * 32, tm = (blockIdx.x / (D / 32)) * 32;
  __shared__ _Float16 tile[32][33];
  const int r = threadIdx.x >> 5, c = threadIdx.x & 31;
  for (int rr = r; rr < 32; rr += 8)
    tile[rr][c] = in[((size_t)b * M + tm + rr) * D + td + c];
  __syncthreads();
  for (int rr = r; rr < 32; rr += 8)
    out[((size_t)b * D + td + rr) * M + tm + c] = tile[c][rr];
}

// ---------------------------------------------------------------------------
// Kernel 4: sim[b,m,n] = dot(txtn[b,m,:], visn[b,n,:]) via f16 WMMA, f32 acc.
// grid (N/64, M/16, B), block 128 = 4 waves. Whole K=256 staged in LDS.
// LDS stride 264 halves (528B): row base steps 16B mod 256B -> conflict-free.
// ---------------------------------------------------------------------------
__global__ __launch_bounds__(128) void sim_gemm(const _Float16* __restrict__ T,
                                                const _Float16* __restrict__ V,
                                                float* __restrict__ S) {
  const int b = blockIdx.z, m0 = blockIdx.y * 16, n0 = blockIdx.x * 64;
  const int tid = threadIdx.x;
  const int wave = tid >> 5, lane = tid & 31, half = lane >> 4, lm = lane & 15;
  __shared__ __align__(16) _Float16 As[16 * 264];
  __shared__ __align__(16) _Float16 Bs[64 * 264];
  const _Float16* Tb = T + ((size_t)b * M + m0) * D;
  const _Float16* Vb = V + ((size_t)b * N + n0) * D;
  for (int idx = tid * 4; idx < 16 * 256; idx += 128 * 4) {
    int r = idx >> 8, c = idx & 255;
    *(v4h*)(&As[r * 264 + c]) = *(const v4h*)(Tb + (size_t)r * D + c);
  }
  for (int idx = tid * 4; idx < 64 * 256; idx += 128 * 4) {
    int r = idx >> 8, c = idx & 255;
    *(v4h*)(&Bs[r * 264 + c]) = *(const v4h*)(Vb + (size_t)r * D + c);
  }
  __syncthreads();
  v8f acc = {};
  const _Float16* ap = &As[lm * 264];
  const _Float16* bp = &Bs[(wave * 16 + lm) * 264];
#pragma unroll
  for (int k0 = 0; k0 < 256; k0 += 32) {
    v16h a = load_afrag(ap + k0, half);
    v16h bb = load_bfrag(bp + k0, half);
    acc = wmma16(a, bb, acc);
  }
  const int n = n0 + wave * 16 + lm;
#pragma unroll
  for (int r = 0; r < 8; r++) {
    int m = m0 + r + 8 * half;
    S[((size_t)b * M + m) * N + n] = acc[r];
  }
}

// ---------------------------------------------------------------------------
// Kernel 5: per (b,m): max/argmax over N, logits, masked maxsim, build pair
// (f16), rewrite masked sim rows to -1.0. One block (256 thr) per row.
// ---------------------------------------------------------------------------
__global__ __launch_bounds__(256) void rowmax_pair(float* __restrict__ S,
                                                   const _Float16* __restrict__ txtn,
                                                   const _Float16* __restrict__ visn,
                                                   const int* __restrict__ mask,
                                                   float* __restrict__ out_maxsim,
                                                   float* __restrict__ out_idx,
                                                   float* __restrict__ out_logits,
                                                   _Float16* __restrict__ pairbuf) {
  const int bm = blockIdx.x, b = bm / M;
  const int t = threadIdx.x;
  const float* row = S + (size_t)bm * N;
  float best = -1e30f; int bidx = 0;
  for (int n = t; n < N; n += 256) {
    float v = row[n];
    if (v > best) { best = v; bidx = n; }
  }
  __shared__ float sv[256]; __shared__ int si[256];
  sv[t] = best; si[t] = bidx;
  __syncthreads();
  for (int s = 128; s > 0; s >>= 1) {
    if (t < s) {
      if (sv[t + s] > sv[t] || (sv[t + s] == sv[t] && si[t + s] < si[t])) {
        sv[t] = sv[t + s]; si[t] = si[t + s];
      }
    }
    __syncthreads();
  }
  const float mx = sv[0];
  const int idx = si[0];
  const int mk = mask[bm];
  if (t == 0) {
    float mm = mk ? mx : 0.0f;
    out_maxsim[bm] = mm;
    out_idx[bm] = (float)idx;
    float pf = fminf(fmaxf(mm, 0.0f), 1.0f);
    out_logits[(size_t)bm * 2 + 0] = 1.0f - pf;
    out_logits[(size_t)bm * 2 + 1] = pf;
  }
  pairbuf[(size_t)bm * 2 * D + t] = txtn[(size_t)bm * D + t];
  pairbuf[(size_t)bm * 2 * D + D + t] = visn[((size_t)b * N + idx) * D + t];
  if (mk == 0) {
    for (int n = t; n < N; n += 256) S[(size_t)bm * N + n] = -1.0f;
  }
}

// ---------------------------------------------------------------------------
// Kernel 6: C[rows x 256] = epi(A[rows x K] @ W + bias), A f16, Wt = W^T f16
// ([256][K], K-contiguous per output col). Block 512 = 16 waves = 16 rows x
// 256 cols (full row in one block enables fused LayerNorm). Output f16.
// EPI: 0 none, 1 GELU, 2 LayerNorm+GELU.
// ---------------------------------------------------------------------------
template <int KDIM, int EPI>
__global__ __launch_bounds__(512) void gemm_rowblock(const _Float16* __restrict__ A,
                                                     const _Float16* __restrict__ Wt,
                                                     const float* __restrict__ bias,
                                                     const float* __restrict__ lng,
                                                     const float* __restrict__ lnb,
                                                     _Float16* __restrict__ Cout) {
  constexpr int STR = KDIM + 8;  // halves; (KDIM+8)*2 mod 256 == 16 -> no conflicts
  const int r0 = blockIdx.x * 16;
  const int tid = threadIdx.x;
  const int wave = tid >> 5, lane = tid & 31, half = lane >> 4, lm = lane & 15;
  __shared__ __align__(16) _Float16 As[16 * STR];
  __shared__ float Cs[16 * 272];
  for (int idx = tid * 4; idx < 16 * KDIM; idx += 512 * 4) {
    int r = idx / KDIM, c = idx % KDIM;
    *(v4h*)(&As[r * STR + c]) = *(const v4h*)(A + (size_t)(r0 + r) * KDIM + c);
  }
  __syncthreads();
  v8f acc = {};
  const int n = wave * 16 + lm;
  const _Float16* ap = &As[lm * STR];
  const _Float16* wp = Wt + (size_t)n * KDIM;
#pragma unroll
  for (int k0 = 0; k0 < KDIM; k0 += 32) {
    v16h a = load_afrag(ap + k0, half);
    v16h bb = load_bfrag(wp + k0, half);
    acc = wmma16(a, bb, acc);
  }
  {
    const float bv = bias[n];
#pragma unroll
    for (int r = 0; r < 8; r++) Cs[(r + 8 * half) * 272 + n] = acc[r] + bv;
  }
  __syncthreads();
  const int row = wave;  // one wave per output row for the epilogue
  if (EPI == 2) {
    float s = 0.f, s2 = 0.f;
    for (int c = lane; c < 256; c += 32) { float v = Cs[row * 272 + c]; s += v; s2 += v * v; }
    for (int off = 16; off > 0; off >>= 1) { s += __shfl_down(s, off, 32); s2 += __shfl_down(s2, off, 32); }
    s = __shfl(s, 0, 32); s2 = __shfl(s2, 0, 32);
    const float mu = s * (1.0f / 256.0f);
    const float var = s2 * (1.0f / 256.0f) - mu * mu;
    const float inv = rsqrtf(var + 1e-5f);
    for (int c = lane; c < 256; c += 32) {
      float v = (Cs[row * 272 + c] - mu) * inv * lng[c] + lnb[c];
      Cs[row * 272 + c] = gelu_exact(v);
    }
  } else if (EPI == 1) {
    for (int c = lane; c < 256; c += 32) Cs[row * 272 + c] = gelu_exact(Cs[row * 272 + c]);
  }
  __syncthreads();
  for (int idx = tid; idx < 16 * 256; idx += 512) {
    int r = idx >> 8, c = idx & 255;
    Cout[(size_t)(r0 + r) * 256 + c] = (_Float16)Cs[r * 272 + c];
  }
}

// ---------------------------------------------------------------------------
// Kernel 7: gate[row] = mask ? dot(g1_row, Wg2)+bg2 : -1e4. 8 rows/block.
// ---------------------------------------------------------------------------
__global__ __launch_bounds__(256) void gate_gemv(const _Float16* __restrict__ A,
                                                 const float* __restrict__ Wg2,
                                                 const float* __restrict__ bg2,
                                                 const int* __restrict__ mask,
                                                 float* __restrict__ gate) {
  const int row = blockIdx.x * 8 + (threadIdx.x >> 5);
  const int lane = threadIdx.x & 31;
  const _Float16* ar = A + (size_t)row * 256;
  float s = 0.f;
  for (int c = lane; c < 256; c += 32) s += (float)ar[c] * Wg2[c];
  for (int off = 16; off > 0; off >>= 1) s += __shfl_down(s, off, 32);
  if (lane == 0) gate[row] = mask[row] ? (s + bg2[0]) : -10000.0f;
}

// ---------------------------------------------------------------------------
// Kernel 8: per-batch finalize: softmax(gate), w, grounding, contradiction,
// score MLP. One block (256 thr) per batch.
// ---------------------------------------------------------------------------
__global__ __launch_bounds__(256) void finalize_batch(const float* __restrict__ gate,
                                                      const int* __restrict__ mask,
                                                      const _Float16* __restrict__ pout,
                                                      const float* __restrict__ maxsim,
                                                      const float* __restrict__ Ws1,
                                                      const float* __restrict__ bs1,
                                                      const float* __restrict__ Ws2,
                                                      const float* __restrict__ bs2,
                                                      float* __restrict__ out_ground,
                                                      float* __restrict__ out_score) {
  const int b = blockIdx.x, t = threadIdx.x;
  __shared__ float red[256];
  __shared__ float wgt[256];
  const float g = gate[(size_t)b * M + t];
  red[t] = g; __syncthreads();
  for (int s = 128; s > 0; s >>= 1) { if (t < s) red[t] = fmaxf(red[t], red[t + s]); __syncthreads(); }
  const float gm = red[0]; __syncthreads();
  const float e = expf(g - gm);
  red[t] = e; __syncthreads();
  for (int s = 128; s > 0; s >>= 1) { if (t < s) red[t] += red[t + s]; __syncthreads(); }
  const float se = red[0]; __syncthreads();
  float wv = (e / se) * (mask[(size_t)b * M + t] ? 1.0f : 0.0f);
  red[t] = wv; __syncthreads();
  for (int s = 128; s > 0; s >>= 1) { if (t < s) red[t] += red[t + s]; __syncthreads(); }
  wv = wv / fmaxf(red[0], 1e-8f); __syncthreads();
  wgt[t] = wv;
  red[t] = maxsim[(size_t)b * M + t] * wv; __syncthreads();
  for (int s = 128; s > 0; s >>= 1) { if (t < s) red[t] += red[t + s]; __syncthreads(); }
  const float contra = 1.0f - red[0]; __syncthreads();
  float acc = 0.f;
  const _Float16* Pb = pout + (size_t)b * M * D;
  for (int m2 = 0; m2 < M; m2++) acc += (float)Pb[(size_t)m2 * D + t] * wgt[m2];
  out_ground[(size_t)b * D + t] = acc;
  const float feat = acc * (1.0f + contra);
  __syncthreads();
  wgt[t] = feat; __syncthreads();
  float hv = 0.f;
  if (t < 128) {
    float a = bs1[t];
    for (int d2 = 0; d2 < 256; d2++) a += wgt[d2] * Ws1[d2 * 128 + t];
    hv = gelu_exact(a);
  }
  red[t] = (t < 128) ? hv * Ws2[t] : 0.0f; __syncthreads();
  for (int s = 128; s > 0; s >>= 1) { if (t < s) red[t] += red[t + s]; __syncthreads(); }
  if (t == 0) out_score[b] = 1.0f / (1.0f + expf(-(red[0] + bs2[0])));
}

// ---------------------------------------------------------------------------
// Kernel 9: weighted_patch[b,n,:] = softmax_m(sim[b,:,n]) @ txtn[b].
// grid (N/16, B), block 512 = 16 waves. Softmax (f32 LDS) -> f16 attn LDS ->
// WMMA against txtnT (B-operand rows K-contiguous in m).
// ---------------------------------------------------------------------------
__global__ __launch_bounds__(512) void patch_kernel(const float* __restrict__ S,
                                                    const _Float16* __restrict__ txtnT,
                                                    float* __restrict__ WP) {
  const int b = blockIdx.y, n0 = blockIdx.x * 16;
  const int tid = threadIdx.x;
  const int wave = tid >> 5, lane = tid & 31, half = lane >> 4, lm = lane & 15;
  __shared__ float Sf[16 * 260];                 // 260: bank = 4*row, conflict-free
  __shared__ __align__(16) _Float16 Ah[16 * 264];
  for (int idx = tid; idx < 16 * 256; idx += 512) {
    int nl = idx & 15, mm = idx >> 4;
    Sf[nl * 260 + mm] = S[((size_t)b * M + mm) * N + n0 + nl];
  }
  __syncthreads();
  {  // softmax over m for row n = wave
    const int row = wave;
    float mx = -1e30f;
    for (int c = lane; c < 256; c += 32) mx = fmaxf(mx, Sf[row * 260 + c]);
    for (int off = 16; off > 0; off >>= 1) mx = fmaxf(mx, __shfl_down(mx, off, 32));
    mx = __shfl(mx, 0, 32);
    float s = 0.f;
    for (int c = lane; c < 256; c += 32) { float e = expf(Sf[row * 260 + c] - mx); Sf[row * 260 + c] = e; s += e; }
    for (int off = 16; off > 0; off >>= 1) s += __shfl_down(s, off, 32);
    s = __shfl(s, 0, 32);
    const float inv = 1.0f / s;
    for (int c = lane; c < 256; c += 32) Ah[row * 264 + c] = (_Float16)(Sf[row * 260 + c] * inv);
  }
  __syncthreads();
  v8f acc = {};
  const _Float16* ap = &Ah[lm * 264];
  const int d = wave * 16 + lm;
  const _Float16* br = txtnT + ((size_t)b * D + d) * M;
#pragma unroll
  for (int k0 = 0; k0 < 256; k0 += 32) {
    v16h a = load_afrag(ap + k0, half);
    v16h bb = load_bfrag(br + k0, half);
    acc = wmma16(a, bb, acc);
  }
#pragma unroll
  for (int r = 0; r < 8; r++) {
    int nn = n0 + r + 8 * half;
    WP[((size_t)b * N + nn) * D + d] = acc[r];
  }
}

// ---------------------------------------------------------------------------
extern "C" void kernel_launch(void* const* d_in, const int* in_sizes, int n_in,
                              void* d_out, int out_size, void* d_ws, size_t ws_size,
                              hipStream_t stream) {
  const float* vis = (const float*)d_in[0];
  const float* txt = (const float*)d_in[1];
  const int* mask = (const int*)d_in[2];
  const float* Wg1 = (const float*)d_in[3];
  const float* bg1 = (const float*)d_in[4];
  const float* Wg2 = (const float*)d_in[5];
  const float* bg2 = (const float*)d_in[6];
  const float* Wo1 = (const float*)d_in[7];
  const float* bo1 = (const float*)d_in[8];
  const float* lng = (const float*)d_in[9];
  const float* lnb = (const float*)d_in[10];
  const float* Wo2 = (const float*)d_in[11];
  const float* bo2 = (const float*)d_in[12];
  const float* Ws1 = (const float*)d_in[13];
  const float* bs1 = (const float*)d_in[14];
  const float* Ws2 = (const float*)d_in[15];
  const float* bs2 = (const float*)d_in[16];

  float* out = (float*)d_out;
  float* out_sim = out;                                // [B,M,N]
  float* out_maxsim = out_sim + (size_t)B * M * N;     // [B,M]
  float* out_idx = out_maxsim + (size_t)B * M;         // [B,M]
  float* out_logits = out_idx + (size_t)B * M;         // [B,M,2]
  float* out_ground = out_logits + (size_t)B * M * 2;  // [B,D]
  float* out_score = out_ground + (size_t)B * D;       // [B,1]
  float* out_wpatch = out_score + (size_t)B;           // [B,N,D]

  char* ws = (char*)d_ws;
  _Float16* visn_h = (_Float16*)ws;                ws += (size_t)B * N * D * 2;
  _Float16* txtn_h = (_Float16*)ws;                ws += (size_t)B * M * D * 2;
  _Float16* txtnT  = (_Float16*)ws;                ws += (size_t)B * M * D * 2;
  _Float16* pair_h = (_Float16*)ws;                ws += (size_t)B * M * 2 * D * 2;
  _Float16* h1_h   = (_Float16*)ws;                ws += (size_t)B * M * D * 2;
  _Float16* pout_h = (_Float16*)ws;                ws += (size_t)B * M * D * 2;
  _Float16* Wt_o1  = (_Float16*)ws;                ws += (size_t)512 * 256 * 2;
  _Float16* Wt_o2  = (_Float16*)ws;                ws += (size_t)256 * 256 * 2;
  _Float16* Wt_g1  = (_Float16*)ws;                ws += (size_t)256 * 256 * 2;
  float* gatebuf   = (float*)ws;                   ws += (size_t)B * M * 4;
  _Float16* g1_h = h1_h;  // alias: h1 dead once pout is computed

  norm_rows_f16<<<B * N, 256, 0, stream>>>(vis, visn_h);
  norm_rows_f16<<<B * M, 256, 0, stream>>>(txt, txtn_h);

  conv_transpose_w<<<(512 * 256) / 256, 256, 0, stream>>>(Wo1, Wt_o1, 512, 256);
  conv_transpose_w<<<(256 * 256) / 256, 256, 0, stream>>>(Wo2, Wt_o2, 256, 256);
  conv_transpose_w<<<(256 * 256) / 256, 256, 0, stream>>>(Wg1, Wt_g1, 256, 256);

  transpose_f16<<<dim3((D / 32) * (M / 32), B), 256, 0, stream>>>(txtn_h, txtnT);

  sim_gemm<<<dim3(N / 64, M / 16, B), 128, 0, stream>>>(txtn_h, visn_h, out_sim);

  rowmax_pair<<<B * M, 256, 0, stream>>>(out_sim, txtn_h, visn_h, mask,
                                         out_maxsim, out_idx, out_logits, pair_h);

  gemm_rowblock<512, 2><<<B * M / 16, 512, 0, stream>>>(pair_h, Wt_o1, bo1, lng, lnb, h1_h);
  gemm_rowblock<256, 0><<<B * M / 16, 512, 0, stream>>>(h1_h, Wt_o2, bo2, lng, lnb, pout_h);
  gemm_rowblock<256, 1><<<B * M / 16, 512, 0, stream>>>(pout_h, Wt_g1, bg1, lng, lnb, g1_h);

  gate_gemv<<<B * M / 8, 256, 0, stream>>>(g1_h, Wg2, bg2, mask, gatebuf);

  finalize_batch<<<B, 256, 0, stream>>>(gatebuf, mask, pout_h, out_maxsim,
                                        Ws1, bs1, Ws2, bs2, out_ground, out_score);

  patch_kernel<<<dim3(N / 16, B), 512, 0, stream>>>(out_sim, txtnT, out_wpatch);
}
